// TGN_67869073211854
// MI455X (gfx1250) — compile-verified, matrix operand order checked by reading, not run on hardware
//
#include <hip/hip_runtime.h>
#include <math.h>

typedef __bf16 bf16;
typedef __bf16 v16bf __attribute__((ext_vector_type(16)));
typedef float  v8f   __attribute__((ext_vector_type(8)));

#define N_NODES 200000        // multiple of 16 and 32
#define BATCH   4096
#define NQ      (3 * BATCH)   // 12288 (mult of 32)
#define KNBR    20
#define DIM     172
#define QD      344
#define KD      516
#define RAWD    688
#define MSGD    100
#define HID     344
#define V8ZERO  {0.f,0.f,0.f,0.f,0.f,0.f,0.f,0.f}

// padded leading dims (K padded to mult of 32)
#define RAWP 704   // 688 -> 704
#define HIDP 352   // 344 -> 352
#define MSGP 128   // 100 -> 128
#define DIMP 192   // 172 -> 192
#define KDP  544   // 516 -> 544
#define QDP  352   // 344 -> 352

// ---------------- bf16 helpers ------------------------------------------------
__device__ __forceinline__ float bf2f(bf16 x) {
  unsigned short u = __builtin_bit_cast(unsigned short, x);
  unsigned int w = ((unsigned int)u) << 16;
  return __builtin_bit_cast(float, w);
}
__device__ __forceinline__ bf16 f2bf(float f) {
  unsigned int u = __builtin_bit_cast(unsigned int, f);
  u += 0x7fffu + ((u >> 16) & 1u);   // round-to-nearest-even
  unsigned short s = (unsigned short)(u >> 16);
  return __builtin_bit_cast(bf16, s);
}
__device__ __forceinline__ void store_elem(float* p, float x) { *p = x; }
__device__ __forceinline__ void store_elem(bf16*  p, float x) { *p = f2bf(x); }

// ---------------- WMMA wrapper ------------------------------------------------
__device__ __forceinline__ v8f wmma_bf16(v16bf a, v16bf b, v8f c) {
  return __builtin_amdgcn_wmma_f32_16x16x32_bf16(false, a, false, b, (short)0, c,
                                                 false, false);
}

// guard-free bf16 fragment loader (operands are padded at their producers).
// ISA 16-bit 16x32 layout: lanes 0-15 hold row (lane&15), K = k0+{0..7,16..23};
// lanes 16-31 same rows, K = k0+{8..15,24..31}.
__device__ __forceinline__ v16bf load_frag_nb(const bf16* __restrict__ base, int ld,
                                              int row0, int k0) {
  const int lane = (int)(threadIdx.x & 31u);
  const bf16* p0 = base + (size_t)(row0 + (lane & 15)) * (size_t)ld
                        + (k0 + ((lane >> 4) << 3));
  v16bf f;
#pragma unroll
  for (int i = 0; i < 8; ++i) f[i] = p0[i];
#pragma unroll
  for (int i = 0; i < 8; ++i) f[i + 8] = p0[16 + i];
  return f;
}

// C tile store (M always a tile multiple; only column guard needed).
// acc[j] -> row = row0 + j + (lane>=16 ? 8 : 0), col = col0 + (lane&15)
template <bool RELU, typename TC>
__device__ __forceinline__ void store_tile(v8f acc, TC* __restrict__ C, int ldc,
                                           int row0, int col0, int Nn,
                                           const float* __restrict__ bias) {
  const int lane = (int)(threadIdx.x & 31u);
  const int c = col0 + (lane & 15);
  if (c >= Nn) return;
  const float bv = bias[c];
  const int rbase = row0 + ((lane >> 4) << 3);
#pragma unroll
  for (int j = 0; j < 8; ++j) {
    float x = acc[j] + bv;
    if (RELU) x = fmaxf(x, 0.f);
    store_elem(&C[(size_t)(rbase + j) * (size_t)ldc + c], x);
  }
}

// ---------------- weight f32 -> zero-padded bf16 arena ------------------------
__global__ void __launch_bounds__(256)
cvt_w_kernel(const float* __restrict__ src, bf16* __restrict__ dst,
             int rows, int cols, int rows_p, int cols_p) {
  const int idx = (int)(blockIdx.x * blockDim.x + threadIdx.x);
  if (idx >= rows_p * cols_p) return;
  const int r = idx / cols_p, c = idx - r * cols_p;
  const float v = (r < rows && c < cols) ? src[(size_t)r * cols + c] : 0.f;
  dst[idx] = f2bf(v);
}

__global__ void __launch_bounds__(256)
zero_bf16_kernel(bf16* __restrict__ p, int n) {
  const int idx = (int)(blockIdx.x * blockDim.x + threadIdx.x);
  if (idx < n) p[idx] = f2bf(0.f);
}

// ---------------- generic 32x32 macro-tile GEMM -------------------------------
// C[M x Nn] = act(A[M x K]bf16 @ W[Nn x K]bf16^T + bias); all K pre-padded.
template <bool RELU, typename TC>
__global__ void __launch_bounds__(256)
gemm_kernel(const bf16* __restrict__ A, int lda, const bf16* __restrict__ W, int ldw,
            const float* __restrict__ bias, TC* __restrict__ C, int ldc,
            int M, int Nn, int Kpad, int tiles_n) {
  const int wid = (int)(blockIdx.x * (blockDim.x >> 5) + (threadIdx.x >> 5));
  const int tm = wid / tiles_n;
  const int tn = wid - tm * tiles_n;
  const int row0 = tm * 32;
  if (row0 >= M) return;
  const int col0 = tn * 32;
  v8f a00 = V8ZERO, a01 = V8ZERO, a10 = V8ZERO, a11 = V8ZERO;
  for (int k0 = 0; k0 < Kpad; k0 += 32) {
    v16bf fa0 = load_frag_nb(A, lda, row0,      k0);
    v16bf fa1 = load_frag_nb(A, lda, row0 + 16, k0);
    v16bf fb0 = load_frag_nb(W, ldw, col0,      k0);
    v16bf fb1 = load_frag_nb(W, ldw, col0 + 16, k0);
    a00 = wmma_bf16(fa0, fb0, a00);
    a01 = wmma_bf16(fa0, fb1, a01);
    a10 = wmma_bf16(fa1, fb0, a10);
    a11 = wmma_bf16(fa1, fb1, a11);
  }
  store_tile<RELU>(a00, C, ldc, row0,      col0,      Nn, bias);
  store_tile<RELU>(a01, C, ldc, row0,      col0 + 16, Nn, bias);
  store_tile<RELU>(a10, C, ldc, row0 + 16, col0,      Nn, bias);
  store_tile<RELU>(a11, C, ldc, row0 + 16, col0 + 16, Nn, bias);
}

// ---------------- kernel 1: fused message MLP ---------------------------------
#define MSG_WAVES 4
#define MSG_WAVE_LDS (16 * RAWP + 16 * HIDP)   // bf16 elems/wave: 11264 + 5632
__global__ void __launch_bounds__(128)
msg_mlp_kernel(const float* __restrict__ raw, const bf16* __restrict__ Wm1b,
               const float* __restrict__ bm1, const bf16* __restrict__ Wm2b,
               const float* __restrict__ bm2, bf16* __restrict__ msg_out) {
  extern __shared__ bf16 smem_dyn[];
  const int widx = (int)(threadIdx.x >> 5);
  const int lane = (int)(threadIdx.x & 31u);
  const int row0 = ((int)blockIdx.x * MSG_WAVES + widx) * 16;   // N mult of 64 tiles
  bf16* sraw = smem_dyn + (size_t)widx * MSG_WAVE_LDS;
  bf16* h1   = sraw + 16 * RAWP;

  // stage raw tile (16 x 688) f32 -> bf16 LDS with zeroed K pad
  for (int e = lane; e < 16 * RAWP; e += 32) {
    const int r = e / RAWP, c = e - r * RAWP;
    const float v = (c < RAWD) ? raw[(size_t)(row0 + r) * RAWD + c] : 0.f;
    sraw[e] = f2bf(v);
  }
  // zero h1 pad cols 344..351
  for (int e = lane; e < 16 * 8; e += 32)
    h1[(e >> 3) * HIDP + HID + (e & 7)] = f2bf(0.f);

  for (int ct = 0; ct < 22; ++ct) {               // h1: 344 cols (+8 pad)
    v8f acc = V8ZERO;
    for (int k0 = 0; k0 < RAWP; k0 += 32) {
      v16bf a = load_frag_nb(sraw, RAWP, 0, k0);
      v16bf b = load_frag_nb(Wm1b, RAWP, ct * 16, k0);
      acc = wmma_bf16(a, b, acc);
    }
    store_tile<true>(acc, h1, HIDP, 0, ct * 16, HID, bm1);
  }
  for (int ct = 0; ct < 7; ++ct) {                // msg: 100 cols (ld 112)
    v8f acc = V8ZERO;
    for (int k0 = 0; k0 < HIDP; k0 += 32) {
      v16bf a = load_frag_nb(h1, HIDP, 0, k0);
      v16bf b = load_frag_nb(Wm2b, HIDP, ct * 16, k0);
      acc = wmma_bf16(a, b, acc);
    }
    store_tile<false>(acc, msg_out, 112, row0, ct * 16, MSGD, bm2);
  }
}

// ---------------- kernel 2: fused GRU memory update ---------------------------
__global__ void __launch_bounds__(32)
gru_kernel(const bf16* __restrict__ msg, const float* __restrict__ memory,
           const bf16* __restrict__ W_ihb, const float* __restrict__ b_ih,
           const bf16* __restrict__ W_hhb, const float* __restrict__ b_hh,
           const int* __restrict__ msg_mask, float* __restrict__ mem_upd) {
  __shared__ float s_rz[16 * 344];    // gi+gh cols 0..343 (r,z gates)
  __shared__ float s_gin[16 * 172];   // gi cols 344..515 (n gate)
  __shared__ float s_ghn[16 * 172];   // gh cols 344..515
  __shared__ bf16  s_msg[16 * MSGP];  // staged msg tile, zero K pad
  __shared__ bf16  s_mem[16 * DIMP];  // staged memory tile, zero K pad
  const int lane = (int)(threadIdx.x & 31u);
  const int row0 = (int)blockIdx.x * 16;   // N_NODES/16 exact

  for (int e = lane; e < 16 * MSGP; e += 32) {
    const int r = e / MSGP, c = e - r * MSGP;
    s_msg[e] = (c < MSGD) ? msg[(size_t)(row0 + r) * 112 + c] : f2bf(0.f);
  }
  for (int e = lane; e < 16 * DIMP; e += 32) {
    const int r = e / DIMP, c = e - r * DIMP;
    const float v = (c < DIM) ? memory[(size_t)(row0 + r) * DIM + c] : 0.f;
    s_mem[e] = f2bf(v);
  }

  for (int ct = 0; ct < 33; ++ct) {   // 516 output cols (+12 pad)
    const int col0 = ct * 16;
    v8f ai = V8ZERO, ah = V8ZERO;
    for (int k0 = 0; k0 < MSGP; k0 += 32) {
      v16bf a = load_frag_nb(s_msg, MSGP, 0, k0);
      v16bf b = load_frag_nb(W_ihb, MSGP, col0, k0);
      ai = wmma_bf16(a, b, ai);
    }
    for (int k0 = 0; k0 < DIMP; k0 += 32) {
      v16bf a = load_frag_nb(s_mem, DIMP, 0, k0);
      v16bf b = load_frag_nb(W_hhb, DIMP, col0, k0);
      ah = wmma_bf16(a, b, ah);
    }
    const int c = col0 + (lane & 15);
    if (c < 3 * DIM) {
      const float bi = b_ih[c], bh = b_hh[c];
      const int rb = (lane >> 4) << 3;
#pragma unroll
      for (int j = 0; j < 8; ++j) {
        const int r = rb + j;
        const float vi = ai[j] + bi, vh = ah[j] + bh;
        if (c < 2 * DIM) s_rz[r * 344 + c] = vi + vh;
        else { s_gin[r * 172 + (c - 2 * DIM)] = vi; s_ghn[r * 172 + (c - 2 * DIM)] = vh; }
      }
    }
  }
  for (int e = lane; e < 16 * DIM; e += 32) {
    const int r = e / DIM, c = e - r * DIM;
    const int node = row0 + r;
    const float rr = 1.f / (1.f + expf(-s_rz[r * 344 + c]));
    const float zz = 1.f / (1.f + expf(-s_rz[r * 344 + c + DIM]));
    const float nn = tanhf(s_gin[r * 172 + c] + rr * s_ghn[r * 172 + c]);
    const float mv = memory[(size_t)node * DIM + c];
    const float hn = (1.f - zz) * nn + zz * mv;
    mem_upd[(size_t)node * DIM + c] = (msg_mask[node] > 0) ? hn : mv;
  }
}

// ---------------- assemble kernels -------------------------------------------
__device__ __forceinline__ int query_node(int n, const int* src, const int* dst,
                                          const int* neg) {
  return (n < BATCH) ? src[n] : ((n < 2 * BATCH) ? dst[n - BATCH] : neg[n - 2 * BATCH]);
}

__global__ void __launch_bounds__(256)
build_qin_kernel(const int* __restrict__ src, const int* __restrict__ dst,
                 const int* __restrict__ neg, const float* __restrict__ mem_upd,
                 const float* __restrict__ node_feats, const float* __restrict__ time_b,
                 bf16* __restrict__ q_in) {
  const int idx = (int)(blockIdx.x * blockDim.x + threadIdx.x);
  if (idx >= NQ * QDP) return;
  const int n = idx / QDP, c = idx - n * QDP;
  float v = 0.f;
  if (c < DIM) {
    const int node = query_node(n, src, dst, neg);
    v = mem_upd[(size_t)node * DIM + c] + node_feats[(size_t)node * DIM + c];
  } else if (c < QD) {
    v = cosf(time_b[c - DIM]);   // t=0 -> cos(b)
  }
  q_in[idx] = f2bf(v);
}

__global__ void __launch_bounds__(256)
build_kin_kernel(const int* __restrict__ nbr, const int* __restrict__ nbr_eidx,
                 const float* __restrict__ edge_times, const float* __restrict__ nbr_times,
                 const float* __restrict__ mem_upd, const float* __restrict__ node_feats,
                 const float* __restrict__ edge_feats, const float* __restrict__ time_w,
                 const float* __restrict__ time_b, bf16* __restrict__ k_in) {
  const int idx = (int)(blockIdx.x * blockDim.x + threadIdx.x);
  if (idx >= NQ * KNBR * KDP) return;
  const int j = idx / KDP, c = idx - j * KDP;
  const int n = j / KNBR;
  float v = 0.f;
  if (c < DIM) {
    const int node = nbr[j];
    v = mem_upd[(size_t)node * DIM + c] + node_feats[(size_t)node * DIM + c];
  } else if (c < 2 * DIM) {
    const int e = nbr_eidx[j];
    v = edge_feats[(size_t)e * DIM + (c - DIM)];
  } else if (c < KD) {
    const int d = c - 2 * DIM;
    const float dt = edge_times[n & (BATCH - 1)] - nbr_times[j];
    v = cosf(dt * time_w[d] + time_b[d]);
  }
  k_in[idx] = f2bf(v);
}

// ---------------- attention: scores + softmax (wave per query) ----------------
__global__ void __launch_bounds__(256)
attn_scores_kernel(const float* __restrict__ q, const bf16* __restrict__ k,
                   const int* __restrict__ nbr, float* __restrict__ attn,
                   int* __restrict__ flags) {
  __shared__ float scs[8][40];
  const int widx = (int)(threadIdx.x >> 5);
  const int lane = (int)(threadIdx.x & 31u);
  const int n = (int)blockIdx.x * 8 + widx;
  if (n >= NQ) return;
  float* sc = scs[widx];
  const float scale = 1.f / sqrtf((float)DIM);
  for (int h = 0; h < 2; ++h) {
    for (int kk = 0; kk < KNBR; ++kk) {
      const size_t j = (size_t)n * KNBR + kk;
      float s = 0.f;
      for (int c = lane; c < DIM; c += 32)
        s += q[(size_t)n * QD + h * DIM + c] * bf2f(k[j * QD + h * DIM + c]);
      for (int off = 16; off > 0; off >>= 1) s += __shfl_down(s, off, 32);
      if (lane == 0) sc[h * KNBR + kk] = (nbr[j] > 0) ? s * scale : -1e9f;
    }
  }
  if (lane == 0) {
    for (int h = 0; h < 2; ++h) {
      float m = -3.4e38f;
      for (int kk = 0; kk < KNBR; ++kk) m = fmaxf(m, sc[h * KNBR + kk]);
      float sum = 0.f;
      for (int kk = 0; kk < KNBR; ++kk) {
        const float e = expf(sc[h * KNBR + kk] - m);
        sc[h * KNBR + kk] = e;
        sum += e;
      }
      const float inv = 1.f / sum;
      for (int kk = 0; kk < KNBR; ++kk)
        attn[(size_t)n * 40 + h * KNBR + kk] = sc[h * KNBR + kk] * inv;
    }
    int any = 0;
    for (int kk = 0; kk < KNBR; ++kk) any |= (nbr[(size_t)n * KNBR + kk] > 0);
    flags[n] = any ? 0 : 1;   // 1 == all neighbors invalid
  }
}

__global__ void __launch_bounds__(256)
ctx_kernel(const float* __restrict__ attn, const bf16* __restrict__ v,
           bf16* __restrict__ ctx) {
  const int idx = (int)(blockIdx.x * blockDim.x + threadIdx.x);
  if (idx >= NQ * QDP) return;
  const int n = idx / QDP, c = idx - n * QDP;
  float s = 0.f;
  if (c < QD) {
    const int h = c / DIM;
#pragma unroll
    for (int kk = 0; kk < KNBR; ++kk)
      s += attn[(size_t)n * 40 + h * KNBR + kk] * bf2f(v[((size_t)n * KNBR + kk) * QD + c]);
  }
  ctx[idx] = f2bf(s);
}

__global__ void __launch_bounds__(256)
build_gin_kernel(const float* __restrict__ attn_out, const int* __restrict__ flags,
                 const int* __restrict__ src, const int* __restrict__ dst,
                 const int* __restrict__ neg, const float* __restrict__ mem_upd,
                 const float* __restrict__ node_feats, bf16* __restrict__ g_in) {
  const int idx = (int)(blockIdx.x * blockDim.x + threadIdx.x);
  if (idx >= NQ * KDP) return;
  const int n = idx / KDP, c = idx - n * KDP;
  float v = 0.f;
  if (c < QD) {
    v = flags[n] ? 0.f : attn_out[(size_t)n * QD + c];
  } else if (c < KD) {
    const int node = query_node(n, src, dst, neg);
    const int d = c - QD;
    v = mem_upd[(size_t)node * DIM + d] + node_feats[(size_t)node * DIM + d];
  }
  g_in[idx] = f2bf(v);
}

// ---------------- launcher ----------------------------------------------------
extern "C" void kernel_launch(void* const* d_in, const int* in_sizes, int n_in,
                              void* d_out, int out_size, void* d_ws, size_t ws_size,
                              hipStream_t stream) {
  (void)in_sizes; (void)n_in; (void)out_size; (void)ws_size;
  const int*   src        = (const int*)d_in[0];
  const int*   dst        = (const int*)d_in[1];
  const int*   neg        = (const int*)d_in[2];
  const int*   nbr        = (const int*)d_in[3];
  const int*   nbr_eidx   = (const int*)d_in[4];
  const int*   msg_mask   = (const int*)d_in[5];
  const float* edge_times = (const float*)d_in[6];
  const float* nbr_times  = (const float*)d_in[7];
  const float* memory     = (const float*)d_in[8];
  const float* node_feats = (const float*)d_in[9];
  const float* edge_feats = (const float*)d_in[10];
  const float* raw_msgs   = (const float*)d_in[11];
  const float* time_w     = (const float*)d_in[12];
  const float* time_b     = (const float*)d_in[13];
  const float* Wm1 = (const float*)d_in[14]; const float* bm1 = (const float*)d_in[15];
  const float* Wm2 = (const float*)d_in[16]; const float* bm2 = (const float*)d_in[17];
  const float* W_ih = (const float*)d_in[18]; const float* b_ih = (const float*)d_in[19];
  const float* W_hh = (const float*)d_in[20]; const float* b_hh = (const float*)d_in[21];
  const float* Wq = (const float*)d_in[22]; const float* bq = (const float*)d_in[23];
  const float* Wk = (const float*)d_in[24]; const float* bk = (const float*)d_in[25];
  const float* Wv = (const float*)d_in[26]; const float* bv = (const float*)d_in[27];
  const float* Wo = (const float*)d_in[28]; const float* bo = (const float*)d_in[29];
  const float* Wg1 = (const float*)d_in[30]; const float* bg1 = (const float*)d_in[31];
  const float* Wg2 = (const float*)d_in[32]; const float* bg2 = (const float*)d_in[33];
  float* out = (float*)d_out;

  // ---- workspace layout (linear, 256B aligned) ----
  char* p = (char*)d_ws;
  size_t off = 0;
  auto carve = [&](size_t bytes) -> void* {
    void* r = (void*)(p + off);
    off = (off + bytes + 255) & ~(size_t)255;
    return r;
  };
  bf16*  msg_bf   = (bf16*) carve((size_t)N_NODES * 112 * sizeof(bf16));
  float* mem_upd  = (float*)carve((size_t)N_NODES * DIM * sizeof(float));
  bf16*  q_in     = (bf16*) carve((size_t)NQ * QDP * sizeof(bf16));
  float* qf       = (float*)carve((size_t)NQ * QD * sizeof(float));
  bf16*  k_in     = (bf16*) carve((size_t)NQ * KNBR * KDP * sizeof(bf16));
  bf16*  kb       = (bf16*) carve((size_t)NQ * KNBR * QD * sizeof(bf16));
  bf16*  vb       = (bf16*) carve((size_t)NQ * KNBR * QD * sizeof(bf16));
  float* attn_w   = (float*)carve((size_t)NQ * 2 * KNBR * sizeof(float));
  int*   flags    = (int*)  carve((size_t)NQ * sizeof(int));
  bf16*  ctx      = (bf16*) carve((size_t)NQ * QDP * sizeof(bf16));
  float* attn_out = (float*)carve((size_t)NQ * QD * sizeof(float));
  bf16*  g_in     = (bf16*) carve((size_t)NQ * KDP * sizeof(bf16));
  bf16*  hbuf     = (bf16*) carve((size_t)NQ * DIMP * sizeof(bf16));
  // zero-padded bf16 weight arena (rows -> tile mult, cols -> mult of 32)
  bf16* Wm1b = (bf16*)carve((size_t)352 * RAWP * sizeof(bf16));
  bf16* Wm2b = (bf16*)carve((size_t)112 * HIDP * sizeof(bf16));
  bf16* Wihb = (bf16*)carve((size_t)528 * MSGP * sizeof(bf16));
  bf16* Whhb = (bf16*)carve((size_t)528 * DIMP * sizeof(bf16));
  bf16* Wqb  = (bf16*)carve((size_t)352 * QDP * sizeof(bf16));
  bf16* Wkb  = (bf16*)carve((size_t)352 * KDP * sizeof(bf16));
  bf16* Wvb  = (bf16*)carve((size_t)352 * KDP * sizeof(bf16));
  bf16* Wob  = (bf16*)carve((size_t)352 * QDP * sizeof(bf16));
  bf16* Wg1b = (bf16*)carve((size_t)192 * KDP * sizeof(bf16));
  bf16* Wg2b = (bf16*)carve((size_t)192 * DIMP * sizeof(bf16));

  auto cvt = [&](const float* s, bf16* d, int rows, int cols, int rows_p, int cols_p) {
    cvt_w_kernel<<<(rows_p * cols_p + 255) / 256, 256, 0, stream>>>(
        s, d, rows, cols, rows_p, cols_p);
  };
  cvt(Wm1, Wm1b, 344, 688, 352, RAWP);
  cvt(Wm2, Wm2b, 100, 344, 112, HIDP);
  cvt(W_ih, Wihb, 516, 100, 528, MSGP);
  cvt(W_hh, Whhb, 516, 172, 528, DIMP);
  cvt(Wq, Wqb, 344, 344, 352, QDP);
  cvt(Wk, Wkb, 344, 516, 352, KDP);
  cvt(Wv, Wvb, 344, 516, 352, KDP);
  cvt(Wo, Wob, 344, 344, 352, QDP);
  cvt(Wg1, Wg1b, 172, 516, 192, KDP);
  cvt(Wg2, Wg2b, 172, 172, 192, DIMP);

  const int row_tiles_N = N_NODES / 16;   // 12500 exact

  // ---- phase 1: node memory update ----
  msg_mlp_kernel<<<row_tiles_N / MSG_WAVES, 128,
                   MSG_WAVES * MSG_WAVE_LDS * sizeof(bf16), stream>>>(
      raw_msgs, Wm1b, bm1, Wm2b, bm2, msg_bf);
  gru_kernel<<<row_tiles_N, 32, 0, stream>>>(
      msg_bf, memory, Wihb, b_ih, Whhb, b_hh, msg_mask, mem_upd);

  // ---- phase 2: assemble q_in / k_in ----
  build_qin_kernel<<<(NQ * QDP + 255) / 256, 256, 0, stream>>>(
      src, dst, neg, mem_upd, node_feats, time_b, q_in);
  build_kin_kernel<<<(NQ * KNBR * KDP + 255) / 256, 256, 0, stream>>>(
      nbr, nbr_eidx, edge_times, nbr_times, mem_upd, node_feats, edge_feats,
      time_w, time_b, k_in);

  // ---- projections (32x32 macro tiles) ----
  {
    const int M = NQ, tn = QDP / 32;                         // 11
    const int waves = (M / 32) * tn;
    gemm_kernel<false, float><<<(waves + 7) / 8, 256, 0, stream>>>(
        q_in, QDP, Wqb, QDP, bq, qf, QD, M, QD, QDP, tn);
  }
  {
    const int M = NQ * KNBR, tn = QDP / 32;                  // 11
    const int waves = (M / 32) * tn;
    gemm_kernel<false, bf16><<<(waves + 7) / 8, 256, 0, stream>>>(
        k_in, KDP, Wkb, KDP, bk, kb, QD, M, QD, KDP, tn);
    gemm_kernel<false, bf16><<<(waves + 7) / 8, 256, 0, stream>>>(
        k_in, KDP, Wvb, KDP, bv, vb, QD, M, QD, KDP, tn);
  }

  // ---- attention ----
  attn_scores_kernel<<<(NQ + 7) / 8, 256, 0, stream>>>(qf, kb, nbr, attn_w, flags);
  ctx_kernel<<<(NQ * QDP + 255) / 256, 256, 0, stream>>>(attn_w, vb, ctx);
  {
    const int M = NQ, tn = QDP / 32;
    const int waves = (M / 32) * tn;
    gemm_kernel<false, float><<<(waves + 7) / 8, 256, 0, stream>>>(
        ctx, QDP, Wob, QDP, bo, attn_out, QD, M, QD, QDP, tn);
  }

  // ---- merge MLP ----
  build_gin_kernel<<<(NQ * KDP + 255) / 256, 256, 0, stream>>>(
      attn_out, flags, src, dst, neg, mem_upd, node_feats, g_in);
  zero_bf16_kernel<<<(NQ * DIMP + 255) / 256, 256, 0, stream>>>(hbuf, NQ * DIMP);
  {
    const int M = NQ, tn = DIMP / 32;                        // 6
    const int waves = (M / 32) * tn;
    gemm_kernel<true, bf16><<<(waves + 7) / 8, 256, 0, stream>>>(
        g_in, KDP, Wg1b, KDP, bg1, hbuf, DIMP, M, DIM, KDP, tn);
  }
  {
    const int M = NQ, tn = DIMP / 32;                        // 6
    const int waves = (M / 32) * tn;
    gemm_kernel<false, float><<<(waves + 7) / 8, 256, 0, stream>>>(
        hbuf, DIMP, Wg2b, DIMP, bg2, out, DIM, M, DIM, DIMP, tn);
  }
}
